// DataTaskGATkLayer_90881507983888
// MI455X (gfx1250) — compile-verified, matrix operand order checked by reading, not run on hardware
//
#include <hip/hip_runtime.h>

typedef __attribute__((ext_vector_type(16))) _Float16 v16h;
typedef __attribute__((ext_vector_type(8)))  _Float16 v8h;
typedef __attribute__((ext_vector_type(4)))  _Float16 v4h;
typedef __attribute__((ext_vector_type(8)))  float    v8f;
typedef __attribute__((ext_vector_type(4)))  float    v4f;
typedef __attribute__((ext_vector_type(2)))  float    v2f;

// ---------------------------------------------------------------------------
// helpers
// ---------------------------------------------------------------------------
__device__ __forceinline__ unsigned enc_f(float f) {
    unsigned u = __float_as_uint(f);
    return (u & 0x80000000u) ? ~u : (u | 0x80000000u);
}
__device__ __forceinline__ float dec_f(unsigned u) {
    return __uint_as_float((u & 0x80000000u) ? (u & 0x7FFFFFFFu) : ~u);
}
__device__ __forceinline__ v16h frag_from(const _Float16* base) {
    v8h lo = *(const v8h*)(base);
    v8h hh = *(const v8h*)(base + 16);
    return __builtin_shufflevector(lo, hh, 0, 1, 2, 3, 4, 5, 6, 7,
                                   8, 9, 10, 11, 12, 13, 14, 15);
}

// ---------------------------------------------------------------------------
// W pre-convert: Wh[col][k] f16 column-major, K zero-padded to Kpad.
// Runs once per GEMM; result is L2-resident for all GEMM blocks, which then
// read B fragments directly via global_load_b128.
// ---------------------------------------------------------------------------
__global__ void convert_w_kernel(const float* __restrict__ W,
                                 _Float16* __restrict__ Wh,
                                 int K, int NOUT, int Kpad) {
    int idx = blockIdx.x * blockDim.x + threadIdx.x;
    if (idx >= NOUT * Kpad) return;
    int col = idx / Kpad, k = idx - col * Kpad;
    Wh[idx] = (k < K) ? (_Float16)W[(long)k * NOUT + col] : (_Float16)0.f;
}

// ---------------------------------------------------------------------------
// WMMA GEMM: Y[N,NOUT] = X[N,K] @ W[K,NOUT] (+ bias). f16 in, f32 accum.
// Block = 128 threads (4 waves); block computes 16 rows x NOUT cols.
// A staged once in LDS (16 x Kpad halves, float4-vectorized); B fragments
// loaded straight from the pre-converted column-major Wh in global memory.
// ---------------------------------------------------------------------------
#define GEMM_BLOCK 128

template <int K, int NOUT>
__global__ void gemm_wmma_kernel(const float* __restrict__ X,
                                 const _Float16* __restrict__ Wh,
                                 const float* __restrict__ bias,
                                 float* __restrict__ Y, int N) {
    constexpr int NK   = (K + 31) / 32;  // K-steps of 32
    constexpr int KPAD = NK * 32;
    constexpr int TPW  = (NOUT / 16) / 4;

    __shared__ _Float16 sA[16 * KPAD];   // [row][k], stride KPAD

    const int tid  = threadIdx.x;
    const int lane = tid & 31;
    const int wave = tid >> 5;
    const int m0   = blockIdx.x * 16;
    const int hi   = (lane >= 16) ? 1 : 0;
    const int ml   = lane & 15;

    // ---- stage A (vectorized: float4 global load -> 4xf16 LDS store) ----
#pragma unroll
    for (int idx4 = tid * 4; idx4 < 16 * KPAD; idx4 += GEMM_BLOCK * 4) {
        int r = idx4 / KPAD, c = idx4 - r * KPAD;
        int rg = m0 + r;
        v4f xv = (v4f){};
        if (rg < N) {
            if (c + 4 <= K) {
                xv = *(const v4f*)&X[(long)rg * K + c];
            } else {
#pragma unroll
                for (int j = 0; j < 4; ++j)
                    if (c + j < K) xv[j] = X[(long)rg * K + c + j];
            }
        }
        v4h hv;
#pragma unroll
        for (int j = 0; j < 4; ++j) hv[j] = (_Float16)xv[j];
        *(v4h*)&sA[idx4] = hv;
    }
    __syncthreads();

    v8f acc[TPW];
#pragma unroll
    for (int t = 0; t < TPW; ++t) acc[t] = (v8f){};

#pragma unroll
    for (int ks = 0; ks < NK; ++ks) {
        v16h afrag = frag_from(&sA[ml * KPAD + ks * 32 + hi * 8]);
#pragma unroll
        for (int t = 0; t < TPW; ++t) {
            const int cb = ((t << 2) + wave) << 4;
            v16h bfrag =
                frag_from(&Wh[(long)(cb + ml) * KPAD + ks * 32 + hi * 8]);
            acc[t] = __builtin_amdgcn_wmma_f32_16x16x32_f16(
                false, afrag, false, bfrag, (short)0, acc[t], false, false);
        }
    }

    // ---- writeback ----
    // D layout: lanes 0-15 -> col=cb+lane, rows m0+v; lanes 16-31 -> rows +8.
    // Fast path (whole 16-row tile in range): one base address per lane,
    // 8 stores with immediate offsets -> compiler can clause them.
    const int rbase = m0 + (hi ? 8 : 0);
#pragma unroll
    for (int t = 0; t < TPW; ++t) {
        const int cb  = ((t << 2) + wave) << 4;
        const int col = cb + ml;
        const float bv = bias ? bias[col] : 0.f;
        float* yb = Y + (long)rbase * NOUT + col;
        if (m0 + 16 <= N) {
#pragma unroll
            for (int v = 0; v < 8; ++v)
                yb[(long)v * NOUT] = acc[t][v] + bv;
        } else {
#pragma unroll
            for (int v = 0; v < 8; ++v)
                if (rbase + v < N) yb[(long)v * NOUT] = acc[t][v] + bv;
        }
    }
}

// ---------------------------------------------------------------------------
// Edge pass 1: z = leaky(xl[src]+xr[dst]+xe, .2); logit[h]=z.att[h].
// One wave per edge; lane owns 4 contiguous channels (b128 gathers).
// Lanes 0-15 cover head 0 (c<64), lanes 16-31 head 1.
// ---------------------------------------------------------------------------
__global__ void edge_logits_kernel(const float* __restrict__ xl,
                                   const float* __restrict__ xr,
                                   const int* __restrict__ src,
                                   const int* __restrict__ dst,
                                   const float* __restrict__ eattr,
                                   const float* __restrict__ We,
                                   const float* __restrict__ att,
                                   float* __restrict__ alog,
                                   unsigned* __restrict__ menc, int E) {
    int gid = blockIdx.x * blockDim.x + threadIdx.x;
    int e = gid >> 5, lane = gid & 31;
    if (e >= E) return;
    int s = src[e], d = dst[e];
    float e0 = eattr[e * 3 + 0], e1 = eattr[e * 3 + 1], e2 = eattr[e * 3 + 2];
    const int c0 = lane * 4;
    v4f xls = *(const v4f*)&xl[(long)s * 128 + c0];
    v4f xrs = *(const v4f*)&xr[(long)d * 128 + c0];
    v4f w0v = *(const v4f*)&We[c0];
    v4f w1v = *(const v4f*)&We[128 + c0];
    v4f w2v = *(const v4f*)&We[256 + c0];
    v4f av  = *(const v4f*)&att[c0];
    float p = 0.f;
#pragma unroll
    for (int j = 0; j < 4; ++j) {
        float xe = e0 * w0v[j] + e1 * w1v[j] + e2 * w2v[j];
        float z = xls[j] + xrs[j] + xe;
        z = z > 0.f ? z : 0.2f * z;
        p += z * av[j];
    }
    // reduce within each 16-lane half (per-head partial sums)
#pragma unroll
    for (int off = 1; off <= 8; off <<= 1) p += __shfl_xor(p, off, 32);
    float other = __shfl_xor(p, 16, 32);   // lane0: head1 total
    if (lane == 0) {
        alog[e * 2 + 0] = p;
        alog[e * 2 + 1] = other;
        atomicMax(&menc[d * 2 + 0], enc_f(p));
        atomicMax(&menc[d * 2 + 1], enc_f(other));
    }
}

// ---------------------------------------------------------------------------
// Edge pass 2: w = exp(logit - max[dst]); num[dst,c] += w*xl[src,c];
// s[dst,h] += w.  One wave per edge, 4 contiguous channels/lane.
// ---------------------------------------------------------------------------
__global__ void edge_scatter_kernel(const float* __restrict__ xl,
                                    const float* __restrict__ alog,
                                    const unsigned* __restrict__ menc,
                                    const int* __restrict__ src,
                                    const int* __restrict__ dst,
                                    float* __restrict__ num,
                                    float* __restrict__ sbuf, int E) {
    int gid = blockIdx.x * blockDim.x + threadIdx.x;
    int e = gid >> 5, lane = gid & 31;
    if (e >= E) return;
    int s = src[e], d = dst[e];
    float w0 = __expf(alog[e * 2 + 0] - dec_f(menc[d * 2 + 0]));
    float w1 = __expf(alog[e * 2 + 1] - dec_f(menc[d * 2 + 1]));
    float w = (lane < 16) ? w0 : w1;
    const int c0 = lane * 4;
    v4f x = *(const v4f*)&xl[(long)s * 128 + c0];
#pragma unroll
    for (int j = 0; j < 4; ++j)
        atomicAdd(&num[(long)d * 128 + c0 + j], w * x[j]);
    if (lane == 0) atomicAdd(&sbuf[d * 2 + 0], w0);
    if (lane == 1) atomicAdd(&sbuf[d * 2 + 1], w1);
}

// ---------------------------------------------------------------------------
// Node finalize: attn = mean over heads (num/s); + residual + bias;
// LayerNorm over 64 channels; leaky 0.01. One wave per node, 2 contiguous
// channels per lane (b64 loads/stores).
// ---------------------------------------------------------------------------
__global__ void node_finalize_kernel(const float* __restrict__ num,
                                     const float* __restrict__ sbuf,
                                     const float* __restrict__ res,
                                     const float* __restrict__ bias,
                                     const float* __restrict__ g,
                                     const float* __restrict__ b,
                                     float* __restrict__ out, int N) {
    int gid = blockIdx.x * blockDim.x + threadIdx.x;
    int i = gid >> 5, lane = gid & 31;
    if (i >= N) return;
    float s0 = sbuf[i * 2 + 0], s1 = sbuf[i * 2 + 1];
    float inv0 = s0 > 0.f ? 1.f / s0 : 0.f;
    float inv1 = s1 > 0.f ? 1.f / s1 : 0.f;
    const int c0 = lane * 2;
    v2f n0 = *(const v2f*)&num[(long)i * 128 + c0];
    v2f n1 = *(const v2f*)&num[(long)i * 128 + 64 + c0];
    v2f rv = *(const v2f*)&res[(long)i * 64 + c0];
    v2f bi = *(const v2f*)&bias[c0];
    float h[2];
#pragma unroll
    for (int j = 0; j < 2; ++j)
        h[j] = 0.5f * (n0[j] * inv0 + n1[j] * inv1) + rv[j] + bi[j];
    float sum = h[0] + h[1];
    float sq  = h[0] * h[0] + h[1] * h[1];
#pragma unroll
    for (int off = 16; off > 0; off >>= 1) {
        sum += __shfl_xor(sum, off, 32);
        sq  += __shfl_xor(sq,  off, 32);
    }
    float mean = sum * (1.f / 64.f);
    float var  = sq * (1.f / 64.f) - mean * mean;
    float rstd = rsqrtf(var + 1e-5f);
    v2f gv = *(const v2f*)&g[c0];
    v2f bv = *(const v2f*)&b[c0];
    v2f yv;
#pragma unroll
    for (int j = 0; j < 2; ++j) {
        float y = (h[j] - mean) * rstd * gv[j] + bv[j];
        yv[j] = y > 0.f ? y : 0.01f * y;
    }
    *(v2f*)&out[(long)i * 64 + c0] = yv;
}

// ---------------------------------------------------------------------------
// Final concat: out[i, :76] = [t1[i, :64], task_x[i, :12]]
// ---------------------------------------------------------------------------
__global__ void concat_kernel(const float* __restrict__ t1,
                              const float* __restrict__ task_x,
                              float* __restrict__ out, int N) {
    int gid = blockIdx.x * blockDim.x + threadIdx.x;
    int total = N * 76;
    if (gid >= total) return;
    int i = gid / 76, c = gid - i * 76;
    out[gid] = (c < 64) ? t1[(long)i * 64 + c] : task_x[(long)i * 12 + (c - 64)];
}

// ---------------------------------------------------------------------------
// host side
// ---------------------------------------------------------------------------
struct LayerP {
    const float *Wl, *bl, *Wr, *br, *We, *att, *bias, *Wres, *g, *b;
};

static LayerP layer_from(void* const* d_in, int base) {
    LayerP p;
    p.Wl   = (const float*)d_in[base + 0];
    p.bl   = (const float*)d_in[base + 1];
    p.Wr   = (const float*)d_in[base + 2];
    p.br   = (const float*)d_in[base + 3];
    p.We   = (const float*)d_in[base + 4];
    p.att  = (const float*)d_in[base + 5];
    p.bias = (const float*)d_in[base + 6];
    p.Wres = (const float*)d_in[base + 7];
    p.g    = (const float*)d_in[base + 8];
    p.b    = (const float*)d_in[base + 9];
    return p;
}

static void launch_gemm(const float* X, const float* W, const float* bias,
                        float* Y, _Float16* wh, int N, int K, int NOUT,
                        hipStream_t stream) {
    int Kpad = ((K + 31) / 32) * 32;
    int nconv = NOUT * Kpad;
    convert_w_kernel<<<dim3((nconv + 255) / 256), dim3(256), 0, stream>>>(
        W, wh, K, NOUT, Kpad);
    dim3 grid((N + 15) / 16), blk(GEMM_BLOCK);
    if (NOUT == 128) {
        if (K == 64)
            gemm_wmma_kernel<64, 128><<<grid, blk, 0, stream>>>(X, wh, bias, Y, N);
        else
            gemm_wmma_kernel<12, 128><<<grid, blk, 0, stream>>>(X, wh, bias, Y, N);
    } else {
        if (K == 64)
            gemm_wmma_kernel<64, 64><<<grid, blk, 0, stream>>>(X, wh, bias, Y, N);
        else
            gemm_wmma_kernel<12, 64><<<grid, blk, 0, stream>>>(X, wh, bias, Y, N);
    }
}

static void run_layer(const float* xsrc, int Ksrc, int Nsrc,
                      const float* xdst, int Kdst, int Ndst,
                      const int* src, const int* dst, const float* eattr, int E,
                      const LayerP& P, _Float16* wh,
                      float* xl, float* xr, float* res, float* num,
                      float* alog, unsigned* menc, float* sbuf,
                      float* out, hipStream_t stream) {
    launch_gemm(xsrc, P.Wl, P.bl, xl, wh, Nsrc, Ksrc, 128, stream);
    launch_gemm(xdst, P.Wr, P.br, xr, wh, Ndst, Kdst, 128, stream);
    launch_gemm(xdst, P.Wres, nullptr, res, wh, Ndst, Kdst, 64, stream);

    hipMemsetAsync(menc, 0, (size_t)Ndst * 2 * sizeof(unsigned), stream);
    hipMemsetAsync(sbuf, 0, (size_t)Ndst * 2 * sizeof(float), stream);
    hipMemsetAsync(num,  0, (size_t)Ndst * 128 * sizeof(float), stream);

    long ethreads = (long)E * 32;
    int  eblocks  = (int)((ethreads + 255) / 256);
    edge_logits_kernel<<<dim3(eblocks), dim3(256), 0, stream>>>(
        xl, xr, src, dst, eattr, P.We, P.att, alog, menc, E);
    edge_scatter_kernel<<<dim3(eblocks), dim3(256), 0, stream>>>(
        xl, alog, menc, src, dst, num, sbuf, E);

    long nthreads = (long)Ndst * 32;
    int  nblocks  = (int)((nthreads + 255) / 256);
    node_finalize_kernel<<<dim3(nblocks), dim3(256), 0, stream>>>(
        num, sbuf, res, P.bias, P.g, P.b, out, Ndst);
}

extern "C" void kernel_launch(void* const* d_in, const int* in_sizes, int n_in,
                              void* d_out, int out_size, void* d_ws, size_t ws_size,
                              hipStream_t stream) {
    const float* task_x = (const float*)d_in[0];   // [NT,12]
    const float* data_x = (const float*)d_in[1];   // [ND,64]
    const int*   dt_ei  = (const int*)d_in[2];     // [2,E]
    const int*   td_ei  = (const int*)d_in[3];     // [2,E]
    const float* eattr  = (const float*)d_in[4];   // [E,3]

    const int NT = in_sizes[0] / 12;
    const int ND = in_sizes[1] / 64;
    const int E  = in_sizes[2] / 2;
    const int NMAX = (NT > ND) ? NT : ND;

    LayerP dt0 = layer_from(d_in, 5);
    LayerP td0 = layer_from(d_in, 15);
    LayerP dt1 = layer_from(d_in, 25);

    // workspace layout (floats)
    float* ws = (float*)d_ws;
    size_t o = 0;
    float*    xl   = ws + o; o += (size_t)NMAX * 128;
    float*    xr   = ws + o; o += (size_t)NMAX * 128;
    float*    res  = ws + o; o += (size_t)NMAX * 64;
    float*    num  = ws + o; o += (size_t)NMAX * 128;
    float*    alog = ws + o; o += (size_t)E * 2;
    unsigned* menc = (unsigned*)(ws + o); o += (size_t)NMAX * 2;
    float*    sbuf = ws + o; o += (size_t)NMAX * 2;
    float*    t0   = ws + o; o += (size_t)NT * 64;
    float*    dbuf = ws + o; o += (size_t)ND * 64;
    float*    t1   = ws + o; o += (size_t)NT * 64;
    _Float16* wh   = (_Float16*)(ws + o); o += (size_t)(128 * 64) / 2;
    (void)ws_size;

    const int* dt_src = dt_ei;       // data indices
    const int* dt_dst = dt_ei + E;   // task indices
    const int* td_src = td_ei;       // task indices
    const int* td_dst = td_ei + E;   // data indices

    // layer dt0: data -> task
    run_layer(data_x, 64, ND, task_x, 12, NT, dt_src, dt_dst, eattr, E, dt0, wh,
              xl, xr, res, num, alog, menc, sbuf, t0, stream);
    // layer td0: task(t0) -> data
    run_layer(t0, 64, NT, data_x, 64, ND, td_src, td_dst, eattr, E, td0, wh,
              xl, xr, res, num, alog, menc, sbuf, dbuf, stream);
    // layer dt1: data(dbuf) -> task(t0)
    run_layer(dbuf, 64, ND, t0, 64, NT, dt_src, dt_dst, eattr, E, dt1, wh,
              xl, xr, res, num, alog, menc, sbuf, t1, stream);

    // concat [t1, task_x]
    long cthreads = (long)NT * 76;
    int  cblocks  = (int)((cthreads + 255) / 256);
    concat_kernel<<<dim3(cblocks), dim3(256), 0, stream>>>(
        t1, task_x, (float*)d_out, NT);
    (void)out_size; (void)n_in;
}
